// GraphBlock_231928234690
// MI455X (gfx1250) — matrix operand order repaired and checked by reading, not compiled, not run on hardware
//
#include <hip/hip_runtime.h>
#include <hip/hip_bf16.h>

// ---------------- problem constants ----------------
constexpr int N_   = 256;
constexpr int P_   = 4;
constexpr int S_   = 40;
constexpr int ENT_ = 60;
constexpr int E_   = 1 + P_ + S_ + ENT_;   // 105
constexpr int D_   = 1024;
constexpr int H_   = 8;
constexpr int T_   = 3;
constexpr int DH_  = D_ / H_;              // 128
constexpr int HT_  = H_ * T_;              // 24
constexpr int EP_  = 112;                  // E padded to mult of 16 (M dim)
constexpr int EK_  = 128;                  // E padded to mult of 32 (K dim)
constexpr float NEG_ = -1e30f;

typedef __bf16 bf16;
typedef __attribute__((ext_vector_type(16))) __bf16 v16bf;
typedef __attribute__((ext_vector_type(8)))  __bf16 v8bf;
typedef __attribute__((ext_vector_type(8)))  float  v8f;

// ---------------- WMMA core ----------------
// One wave computes a 16x64 f32 tile (4 accumulators sharing one A fragment
// per k-step -> 2.5 b128 loads per v_wmma instead of 4).
// A: row-major (16 rows x K), lda elements.  Bt: B stored column-major, i.e.
// row n of Bt is output-column n, contiguous over K, ldb elements.
// Fragment layouts per CDNA5 ISA 7.12.2 (wave32):
//   A 16x32 bf16 : lanes 0-15 -> K[0..7],K[16..23]; lanes16-31 -> K[8..15],K[24..31]; row = lane%16
//   B 32x16 bf16 : lanes 0-15 -> K[0..7],K[16..23]; lanes16-31 -> K[8..15],K[24..31]; col = lane%16
__device__ __forceinline__ v16bf cmb(v8bf a0, v8bf a1) {
  v16bf r;
#pragma unroll
  for (int i = 0; i < 8; ++i) { r[i] = a0[i]; r[i + 8] = a1[i]; }
  return r;
}

// 4 N-tiles (64 output columns) per wave
__device__ __forceinline__ void wmma_tile4(const bf16* A, int lda,
                                           const bf16* Bt, int ldb,
                                           int ksteps, v8f c[4]) {
  const int lane = threadIdx.x & 31;
  const int half = lane >> 4;     // 0 or 1
  const int r16  = lane & 15;
  const bf16* arow = A  + (long)r16 * lda;
  const bf16* brow = Bt + (long)r16 * ldb;
  for (int ks = 0; ks < ksteps; ++ks) {
    const int kk = ks * 32;
    v8bf a0 = *(const v8bf*)(arow + kk + half * 8);
    v8bf a1 = *(const v8bf*)(arow + kk + 16 + half * 8);
    v16bf af = cmb(a0, a1);
#pragma unroll
    for (int t = 0; t < 4; ++t) {
      const bf16* bp = brow + (long)t * 16 * ldb + kk;
      v8bf b0 = *(const v8bf*)(bp + half * 8);
      v8bf b1 = *(const v8bf*)(bp + 16 + half * 8);
      c[t] = __builtin_amdgcn_wmma_f32_16x16x32_bf16(
          false, af, false, cmb(b0, b1), (short)0, c[t], false, false);
    }
  }
}

// ---------------- prep kernels ----------------
__global__ void build_xb(const float* __restrict__ qv, const float* __restrict__ para,
                         const float* __restrict__ sent, const float* __restrict__ ent,
                         bf16* __restrict__ xb) {
  long idx = (long)blockIdx.x * 256 + threadIdx.x;
  if (idx >= (long)N_ * E_ * D_) return;
  int d = (int)(idx % D_);
  long r = idx / D_;
  int e = (int)(r % E_);
  int n = (int)(r / E_);
  float x;
  if (e == 0)            x = qv[(long)n * D_ + d];
  else if (e <= P_)      x = para[((long)n * P_ + (e - 1)) * D_ + d];
  else if (e <= P_ + S_) x = sent[((long)n * S_ + (e - 1 - P_)) * D_ + d];
  else                   x = ent [((long)n * ENT_ + (e - 1 - P_ - S_)) * D_ + d];
  xb[idx] = (bf16)x;
}

__global__ void f2bf(const float* __restrict__ in, bf16* __restrict__ out, long count) {
  long idx = (long)blockIdx.x * 256 + threadIdx.x;
  if (idx < count) out[idx] = (bf16)in[idx];
}

// out[b][c][r] = in[b][r][c], converted to bf16
__global__ void trans_f2bf(const float* __restrict__ in, bf16* __restrict__ out,
                           int R, int C, long total) {
  long idx = (long)blockIdx.x * 256 + threadIdx.x;
  if (idx >= total) return;
  int c = (int)(idx % C);
  long t = idx / C;
  int r = (int)(t % R);
  long b = t / R;
  out[(b * C + c) * (long)R + r] = (bf16)in[idx];
}

__global__ void relu_bf(const float* __restrict__ in, bf16* __restrict__ out, long count) {
  long idx = (long)blockIdx.x * 256 + threadIdx.x;
  if (idx < count) out[idx] = (bf16)fmaxf(in[idx], 0.f);
}

// g = sigmoid(gtmp); gg1 = g[:,:DH]*a1 ; gg2 = g[:,DH:]*a2   (bf16)
__global__ void sig_split(const float* __restrict__ g, const float* __restrict__ a,
                          bf16* __restrict__ gg1, bf16* __restrict__ gg2) {
  long idx = (long)blockIdx.x * 256 + threadIdx.x;
  if (idx >= (long)HT_ * N_ * 2 * DH_) return;
  int k = (int)(idx % (2 * DH_));
  long t = idx / (2 * DH_);
  int n  = (int)(t % N_);
  int ht = (int)(t / N_);
  float gv = 1.f / (1.f + __expf(-g[idx]));
  float av = a[(long)ht * 2 * DH_ + k];
  bf16 val = (bf16)(gv * av);
  if (k < DH_) gg1[((long)ht * N_ + n) * DH_ + k]         = val;
  else         gg2[((long)ht * N_ + n) * DH_ + (k - DH_)] = val;
}

// ---------------- generic batched WMMA GEMM: C = A @ B (fp32 out) ----------------
// grid: (M/16, N/64, batch), 32 threads
__global__ void gemm_generic(const bf16* __restrict__ A, int lda, long sA,
                             const bf16* __restrict__ Bt, int ldb, long sB,
                             float* __restrict__ C, int ldc, long sC, int ksteps) {
  const int b = blockIdx.z;
  const bf16* Ab = A  + b * sA + (long)blockIdx.x * 16 * lda;
  const bf16* Bb = Bt + b * sB + (long)blockIdx.y * 64 * ldb;
  v8f c[4] = {};
  wmma_tile4(Ab, lda, Bb, ldb, ksteps, c);
  const int lane = threadIdx.x & 31;
  float* Cb = C + b * sC;
  const int col0 = blockIdx.y * 64 + (lane & 15);
  const int row0 = blockIdx.x * 16 + ((lane >> 4) << 3);
#pragma unroll
  for (int t = 0; t < 4; ++t)
#pragma unroll
    for (int v = 0; v < 8; ++v)
      Cb[(long)(row0 + v) * ldc + col0 + t * 16] = c[t][v];
}

// ---------------- h_last GEMM: (N*E x 1024) @ W[h,T-1] -> masked, transposed bf16 ----------------
// hlastT[h][n][dcol][e]  (e-dim padded to EK_);  grid ((N*E)/16, DH/64, H)
__global__ void gemm_hlast(const bf16* __restrict__ xb, const bf16* __restrict__ Wtb,
                           const float* __restrict__ pmask, const float* __restrict__ smask,
                           const float* __restrict__ emask, bf16* __restrict__ hlastT) {
  const int h = blockIdx.z;
  const bf16* A  = xb + (long)blockIdx.x * 16 * D_;
  const bf16* Bt = Wtb + ((long)(h * T_ + (T_ - 1))) * DH_ * D_ + (long)blockIdx.y * 64 * D_;
  v8f c[4] = {};
  wmma_tile4(A, D_, Bt, D_, D_ / 32, c);
  const int lane = threadIdx.x & 31;
  const int d0 = blockIdx.y * 64 + (lane & 15);
  const int m0 = blockIdx.x * 16 + ((lane >> 4) << 3);
#pragma unroll
  for (int v = 0; v < 8; ++v) {
    int m = m0 + v;
    int n = m / E_;
    int e = m - n * E_;
    float mask;
    if (e == 0)            mask = 1.f;
    else if (e <= P_)      mask = pmask[n * P_ + (e - 1)];
    else if (e <= P_ + S_) mask = smask[n * S_ + (e - 1 - P_)];
    else                   mask = emask[n * ENT_ + (e - 1 - P_ - S_)];
    bf16* dst = hlastT + ((long)h * N_ + n) * DH_ * EK_ + e;
#pragma unroll
    for (int t = 0; t < 4; ++t)
      dst[(long)(d0 + t * 16) * EK_] = (bf16)(c[t][v] * mask);
  }
}

// ---------------- f1/f2: batched dot products, wave-parallel over D ----------------
__global__ void f_kernel(const bf16* __restrict__ xb, const float* __restrict__ Wg1,
                         const float* __restrict__ Wg2, float* __restrict__ f1,
                         float* __restrict__ f2) {
  __shared__ float w1[D_], w2[D_];
  const int blk = blockIdx.x;          // ht*N + n
  const int n = blk % N_, ht = blk / N_;
  const float* W1 = Wg1 + ((long)ht * N_ + n) * D_;
  const float* W2 = Wg2 + ((long)ht * N_ + n) * D_;
  for (int i = threadIdx.x; i < D_; i += 256) { w1[i] = W1[i]; w2[i] = W2[i]; }
  __syncthreads();
  const int wave = threadIdx.x >> 5, lane = threadIdx.x & 31;
  for (int e = wave; e < E_; e += 8) {
    const bf16* xr = xb + ((long)n * E_ + e) * D_;
    float a1 = 0.f, a2 = 0.f;
    for (int d = lane * 2; d < D_; d += 64) {
      float x0 = (float)xr[d], x1 = (float)xr[d + 1];
      a1 += x0 * w1[d] + x1 * w1[d + 1];
      a2 += x0 * w2[d] + x1 * w2[d + 1];
    }
#pragma unroll
    for (int off = 16; off; off >>= 1) {
      a1 += __shfl_xor(a1, off);
      a2 += __shfl_xor(a2, off);
    }
    if (lane == 0) {
      f1[((long)ht * N_ + n) * E_ + e] = a1;
      f2[((long)ht * N_ + n) * E_ + e] = a2;
    }
  }
}

// ---------------- edge-type select + leaky-relu + mask + softmax -> bf16 coefs ----------------
__global__ void softmax_kernel(const float* __restrict__ f1, const float* __restrict__ f2,
                               const int* __restrict__ adj, bf16* __restrict__ coefsb) {
  __shared__ float red[128];
  const int blk = blockIdx.x;              // (h*N + n)*E + i
  const int i = blk % E_;
  const int t2 = blk / E_;
  const int n = t2 % N_, h = t2 / N_;
  const int j = threadIdx.x;               // 0..127
  float f1v[T_];
#pragma unroll
  for (int t = 0; t < T_; ++t) f1v[t] = f1[((long)(h * T_ + t) * N_ + n) * E_ + i];
  const bool valid = j < E_;
  float s = NEG_;
  if (valid) {
    int a = adj[((long)n * E_ + i) * E_ + j];
    if (a > 0) {
      int t = a - 1;
      float v = f1v[t] + f2[((long)(h * T_ + t) * N_ + n) * E_ + j];
      s = v > 0.f ? v : 0.2f * v;          // leaky_relu(., 0.2)
    }
  }
  red[j] = s; __syncthreads();
  for (int st = 64; st; st >>= 1) { if (j < st) red[j] = fmaxf(red[j], red[j + st]); __syncthreads(); }
  const float m = red[0]; __syncthreads();
  const float ex = valid ? __expf(s - m) : 0.f;
  red[j] = ex; __syncthreads();
  for (int st = 64; st; st >>= 1) { if (j < st) red[j] += red[j + st]; __syncthreads(); }
  const float inv = 1.f / red[0];
  coefsb[(((long)h * N_ + n) * EP_ + i) * EK_ + j] = (bf16)(valid ? ex * inv : 0.f);
}

// ---------------- final GEMM: coefs @ h_last, fused relu + fp32 residual ----------------
// grid (EP/16, DH/64, H*N)
__global__ void gemm_out(const bf16* __restrict__ coefsb, const bf16* __restrict__ hlastT,
                         const float* __restrict__ qv, const float* __restrict__ para,
                         const float* __restrict__ sent, const float* __restrict__ ent,
                         float* __restrict__ out) {
  const int z = blockIdx.z;
  const int h = z >> 8, n = z & 255;
  const bf16* A  = coefsb + (((long)h * N_ + n) * EP_ + blockIdx.x * 16) * (long)EK_;
  const bf16* Bt = hlastT + (((long)h * N_ + n) * DH_ + blockIdx.y * 64) * (long)EK_;
  v8f c[4] = {};
  wmma_tile4(A, EK_, Bt, EK_, EK_ / 32, c);
  const int lane = threadIdx.x & 31;
  const int d0 = blockIdx.y * 64 + (lane & 15);
  const int i0 = blockIdx.x * 16 + ((lane >> 4) << 3);
#pragma unroll
  for (int v = 0; v < 8; ++v) {
    int i = i0 + v;
    if (i >= E_) continue;                 // padded rows discarded
    const float* xrow;
    if (i == 0)            xrow = qv   + (long)n * D_;
    else if (i <= P_)      xrow = para + ((long)n * P_ + (i - 1)) * D_;
    else if (i <= P_ + S_) xrow = sent + ((long)n * S_ + (i - 1 - P_)) * D_;
    else                   xrow = ent  + ((long)n * ENT_ + (i - 1 - P_ - S_)) * D_;
    float* orow = out + ((long)n * E_ + i) * D_ + h * DH_;
#pragma unroll
    for (int t = 0; t < 4; ++t) {
      int d = d0 + t * 16;
      orow[d] = fmaxf(c[t][v], 0.f) + xrow[h * DH_ + d];
    }
  }
}

// ---------------- host launcher ----------------
extern "C" void kernel_launch(void* const* d_in, const int* in_sizes, int n_in,
                              void* d_out, int out_size, void* d_ws, size_t ws_size,
                              hipStream_t stream) {
  const float* qv    = (const float*)d_in[0];
  const float* para  = (const float*)d_in[1];
  const float* sent  = (const float*)d_in[2];
  const float* ent   = (const float*)d_in[3];
  const float* pmask = (const float*)d_in[4];
  const float* smask = (const float*)d_in[5];
  const float* emask = (const float*)d_in[6];
  const int*   adj   = (const int*)d_in[7];
  const float* W     = (const float*)d_in[8];
  const float* a     = (const float*)d_in[9];
  const float* qW1   = (const float*)d_in[10];
  const float* qW2   = (const float*)d_in[11];
  float* out = (float*)d_out;

  // workspace carve-out (256B aligned)
  char* base = (char*)d_ws;
  size_t off = 0;
  auto alloc = [&](size_t bytes) -> char* {
    char* p = base + off;
    off = (off + bytes + 255) & ~(size_t)255;
    return p;
  };
  bf16*  xb     = (bf16*) alloc((size_t)N_ * E_ * D_ * 2);
  bf16*  Wb     = (bf16*) alloc((size_t)HT_ * D_ * DH_ * 2);
  bf16*  Wtb    = (bf16*) alloc((size_t)HT_ * DH_ * D_ * 2);
  bf16*  qW1t   = (bf16*) alloc((size_t)HT_ * 2 * DH_ * D_ * 2);
  bf16*  qW2t   = (bf16*) alloc((size_t)HT_ * 2 * DH_ * 2 * DH_ * 2);
  float* qtmp   = (float*)alloc((size_t)HT_ * N_ * 2 * DH_ * 4);   // reused as gtmp
  bf16*  q1b    = (bf16*) alloc((size_t)HT_ * N_ * 2 * DH_ * 2);
  bf16*  gg1b   = (bf16*) alloc((size_t)HT_ * N_ * DH_ * 2);
  bf16*  gg2b   = (bf16*) alloc((size_t)HT_ * N_ * DH_ * 2);
  float* Wg1    = (float*)alloc((size_t)HT_ * N_ * D_ * 4);
  float* Wg2    = (float*)alloc((size_t)HT_ * N_ * D_ * 4);
  float* f1     = (float*)alloc((size_t)HT_ * N_ * E_ * 4);
  float* f2     = (float*)alloc((size_t)HT_ * N_ * E_ * 4);
  bf16*  coefsb = (bf16*) alloc((size_t)H_ * N_ * EP_ * EK_ * 2);
  bf16*  hlastT = (bf16*) alloc((size_t)H_ * N_ * DH_ * EK_ * 2);
  (void)ws_size; (void)in_sizes; (void)n_in; (void)out_size;

  auto blks = [](long n) { return (unsigned)((n + 255) / 256); };

  // --- prep: bf16 conversions / transposes ---
  build_xb<<<blks((long)N_ * E_ * D_), 256, 0, stream>>>(qv, para, sent, ent, xb);
  f2bf<<<blks((long)HT_ * D_ * DH_), 256, 0, stream>>>(W, Wb, (long)HT_ * D_ * DH_);
  trans_f2bf<<<blks((long)HT_ * D_ * DH_), 256, 0, stream>>>(W, Wtb, D_, DH_, (long)HT_ * D_ * DH_);
  trans_f2bf<<<blks((long)HT_ * D_ * 2 * DH_), 256, 0, stream>>>(qW1, qW1t, D_, 2 * DH_, (long)HT_ * D_ * 2 * DH_);
  trans_f2bf<<<blks((long)HT_ * 2 * DH_ * 2 * DH_), 256, 0, stream>>>(qW2, qW2t, 2 * DH_, 2 * DH_, (long)HT_ * 2 * DH_ * 2 * DH_);

  // --- gate path: q1 = relu(qv @ qW1);  g = sigmoid(q1 @ qW2);  gg = g * a ---
  // A = query rows inside xb (e==0 rows), effective lda = E_*D_
  gemm_generic<<<dim3(N_ / 16, (2 * DH_) / 64, HT_), 32, 0, stream>>>(
      xb, E_ * D_, 0L, qW1t, D_, (long)2 * DH_ * D_, qtmp, 2 * DH_, (long)N_ * 2 * DH_, D_ / 32);
  relu_bf<<<blks((long)HT_ * N_ * 2 * DH_), 256, 0, stream>>>(qtmp, q1b, (long)HT_ * N_ * 2 * DH_);
  gemm_generic<<<dim3(N_ / 16, (2 * DH_) / 64, HT_), 32, 0, stream>>>(
      q1b, 2 * DH_, (long)N_ * 2 * DH_, qW2t, 2 * DH_, (long)2 * DH_ * 2 * DH_,
      qtmp, 2 * DH_, (long)N_ * 2 * DH_, (2 * DH_) / 32);
  sig_split<<<blks((long)HT_ * N_ * 2 * DH_), 256, 0, stream>>>(qtmp, a, gg1b, gg2b);

  // --- Wg1/Wg2 = gg @ W^T  (B col-major == original W layout) ---
  gemm_generic<<<dim3(N_ / 16, D_ / 64, HT_), 32, 0, stream>>>(
      gg1b, DH_, (long)N_ * DH_, Wb, DH_, (long)D_ * DH_, Wg1, D_, (long)N_ * D_, DH_ / 32);
  gemm_generic<<<dim3(N_ / 16, D_ / 64, HT_), 32, 0, stream>>>(
      gg2b, DH_, (long)N_ * DH_, Wb, DH_, (long)D_ * DH_, Wg2, D_, (long)N_ * D_, DH_ / 32);

  // --- f1/f2 attention logits ---
  f_kernel<<<HT_ * N_, 256, 0, stream>>>(xb, Wg1, Wg2, f1, f2);

  // --- h_last = (x @ W[:,T-1]) * mask, stored transposed bf16 ---
  gemm_hlast<<<dim3((N_ * E_) / 16, DH_ / 64, H_), 32, 0, stream>>>(
      xb, Wtb, pmask, smask, emask, hlastT);

  // --- softmax over edges -> bf16 coefs (K-padded with zeros) ---
  softmax_kernel<<<H_ * N_ * E_, 128, 0, stream>>>(f1, f2, adj, coefsb);

  // --- out = relu(coefs @ h_last) + x  (fp32 residual from original inputs) ---
  gemm_out<<<dim3(EP_ / 16, DH_ / 64, H_ * N_), 32, 0, stream>>>(
      coefsb, hlastT, qv, para, sent, ent, out);
}
